// MultiLevelMultiHeadedAttention_64415919506137
// MI455X (gfx1250) — matrix operand order repaired and checked by reading, not compiled, not run on hardware
//
#include <hip/hip_runtime.h>
#include <hip/hip_bf16.h>

// ---------------- problem constants ----------------
#define BB   16
#define NN   4
#define T1C  256
#define T2C  1024
#define FF   512
#define HH   8
#define DKC  64
#define SCALE_QK 0.125f   // 1/sqrt(64)

typedef __bf16  v16bf __attribute__((ext_vector_type(16)));
typedef __bf16  v4bf  __attribute__((ext_vector_type(4)));
typedef float   v8f   __attribute__((ext_vector_type(8)));

__device__ __forceinline__ v8f wmma_bf16(v16bf a, v16bf b, v8f c) {
  return __builtin_amdgcn_wmma_f32_16x16x32_bf16(false, a, false, b, (short)0, c, false, false);
}

// generic -> LDS offset (addrspace(3) cast truncates to the 32-bit LDS offset)
__device__ __forceinline__ uint32_t lds_off(const void* p) {
  return (uint32_t)(uintptr_t)(__attribute__((address_space(3))) const void*)p;
}

// ---- bf16 fragment loaders (CDNA5 wave32 16x16x32 layouts) ----
// A 16x32 (MxK): lane m=lane&15, kg=lane>>4; elems 0..7 -> K=kg*8+i, 8..15 -> K=16+kg*8+i
__device__ __forceinline__ v16bf load_a_bf(const __bf16* A, int lda, int row0, int k0, int lane) {
  const int m = lane & 15, kg = lane >> 4;
  const __bf16* p = A + (size_t)(row0 + m) * lda + k0 + kg * 8;
  v16bf r;
  ((uint4*)&r)[0] = *(const uint4*)p;        // 8 bf16
  ((uint4*)&r)[1] = *(const uint4*)(p + 16); // 8 bf16
  return r;
}
// B 32x16 (KxN) from row-major source S: frag column n = source row, K = source col.
// lane n=lane&15, kg=lane>>4; elems 0..7 -> K=kg*8+i, 8..15 -> K=16+kg*8+i
__device__ __forceinline__ v16bf load_bT_bf(const __bf16* S, int lds_, int nrow0, int k0, int lane) {
  const int n = lane & 15, kg = lane >> 4;
  const __bf16* p = S + (size_t)(nrow0 + n) * lds_ + k0 + kg * 8;
  v16bf r;
  ((uint4*)&r)[0] = *(const uint4*)p;
  ((uint4*)&r)[1] = *(const uint4*)(p + 16);
  return r;
}

// ---------------- fp32 -> bf16 pack (one-time, bandwidth-bound) ----------------
__global__ __launch_bounds__(256)
void pack_bf16(const float* __restrict__ src, __bf16* __restrict__ dst, int n4) {
  const int i = blockIdx.x * blockDim.x + threadIdx.x;
  if (i >= n4) return;
  const float4 v = ((const float4*)src)[i];
  v4bf o;
  o[0] = (__bf16)v.x; o[1] = (__bf16)v.y; o[2] = (__bf16)v.z; o[3] = (__bf16)v.w;
  ((v4bf*)dst)[i] = o;
}

// ---------------- GEMM: C(M,N) = A(M,K) * W(N,K)^T + bias ----------------
// A, W pre-packed bf16. 8 waves/block, each wave -> 32(M) x 64(N) tile:
// 2 A-frags x 4 B-frags -> 8 WMMAs per K=32 step.
__global__ __launch_bounds__(256)
void gemm_bf16_wmma(const __bf16* __restrict__ A, const __bf16* __restrict__ W,
                    const float* __restrict__ bias, float* __restrict__ C,
                    __bf16* __restrict__ Cbf, float bf_scale,
                    int M, int N, int K, int vtT2) {
  const int lane = threadIdx.x & 31;
  const int wave = threadIdx.x >> 5;
  const int m16  = lane & 15, half = lane >> 4;
  const int row0 = blockIdx.x * 32;
  const int col0 = wave * 64;

  v8f acc[2][4] = {};
  for (int k0 = 0; k0 < K; k0 += 32) {
    const v16bf va0 = load_a_bf(A, K, row0,      k0, lane);
    const v16bf va1 = load_a_bf(A, K, row0 + 16, k0, lane);
#pragma unroll
    for (int j = 0; j < 4; ++j) {
      const v16bf vb = load_bT_bf(W, K, col0 + j * 16, k0, lane);
      acc[0][j] = wmma_bf16(va0, vb, acc[0][j]);
      acc[1][j] = wmma_bf16(va1, vb, acc[1][j]);
    }
  }
#pragma unroll
  for (int t = 0; t < 2; ++t) {
#pragma unroll
    for (int j = 0; j < 4; ++j) {
      const int col = col0 + j * 16 + m16;
      const float bcol = bias[col];
#pragma unroll
      for (int r = 0; r < 8; ++r) {
        const int row = row0 + t * 16 + r + 8 * half;
        const float val = acc[t][j][r] + bcol;
        if (C) C[(size_t)row * N + col] = val;
        if (Cbf) {
          const float v = val * bf_scale;
          if (vtT2 > 0) {  // (bn, h, dk, T2) transposed layout for V
            const int bn = row / vtT2, t2 = row % vtT2;
            const int hh = col >> 6, dk = col & 63;
            Cbf[(((size_t)bn * (N >> 6) + hh) * 64 + dk) * vtT2 + t2] = (__bf16)v;
          } else {
            Cbf[(size_t)row * N + col] = (__bf16)v;
          }
        }
      }
    }
  }
}

// ---------------- Stage 1: per-(b,n,h) flash attention over T2 ----------------
// K/V tiles staged into LDS once per block via async-to-LDS DMA (double-buffered),
// shared by all 16 waves (16x traffic reduction vs per-wave global loads).
__global__ __launch_bounds__(512)
void attn_stage1(const __bf16* __restrict__ qb,   // (B*T1, F), pre-scaled by 1/sqrt(DK)
                 const __bf16* __restrict__ kb,   // (B*N*T2, F)
                 const __bf16* __restrict__ vt,   // (B*N*H, DK, T2)
                 const int* __restrict__ mask_audio, // (B*N, T2)
                 float* __restrict__ phone)       // (B*N*H, T1, DK)
{
  __shared__ __align__(16) __bf16 kbuf[2][32 * 64];   // K tile: rows t2local, cols dk
  __shared__ __align__(16) __bf16 vbuf[2][64 * 32];   // V tile: rows dk, cols t2local
  __shared__ __align__(16) __bf16 plds[16][16 * 32];  // per-wave P bounce
  const int tid  = threadIdx.x;
  const int lane = tid & 31;
  const int wave = tid >> 5;
  const int m16  = lane & 15, half = lane >> 4;
  const int bnh = blockIdx.x;
  const int h  = bnh & (HH - 1);
  const int bn = bnh >> 3;
  const int b  = bn >> 2;
  const int t1_0 = wave * 16;

  const __bf16* qrow  = qb + (size_t)b * T1C * FF;
  const __bf16* kbase = kb + (size_t)bn * T2C * FF;
  const __bf16* vbase = vt + (size_t)bnh * DKC * T2C;
  const int*    ma    = mask_audio + (size_t)bn * T2C;

  const v16bf aq0 = load_a_bf(qrow, FF, t1_0, h * 64, lane);
  const v16bf aq1 = load_a_bf(qrow, FF, t1_0, h * 64 + 32, lane);

  v8f acc[4] = {};
  float rmax[8], rsum[8];
#pragma unroll
  for (int r = 0; r < 8; ++r) { rmax[r] = -__builtin_inff(); rsum[r] = 0.f; }

  // each of the 512 threads DMAs one 16B chunk per tile-pair:
  //   tid < 256 : K tile 32x64 (8 chunks/row), tid >= 256 : V tile 64x32 (4 chunks/row)
  const int kr = tid >> 3,          kc = (tid & 7) * 8;   // K: row, col
  const int vr = (tid - 256) >> 2,  vc = (tid & 3) * 8;   // V: row(dk), col

  auto issue_tile = [&](int buf, int t2) {
    if (tid < 256) {
      const __bf16* g = kbase + (size_t)(t2 + kr) * FF + h * 64 + kc;
      const uint32_t l = lds_off(&kbuf[buf][kr * 64 + kc]);
      asm volatile("global_load_async_to_lds_b128 %0, %1, off" :: "v"(l), "v"(g) : "memory");
    } else {
      const __bf16* g = vbase + (size_t)vr * T2C + t2 + vc;
      const uint32_t l = lds_off(&vbuf[buf][vr * 32 + vc]);
      asm volatile("global_load_async_to_lds_b128 %0, %1, off" :: "v"(l), "v"(g) : "memory");
    }
  };

  issue_tile(0, 0);  // prologue

  for (int t2 = 0; t2 < T2C; t2 += 32) {
    const int buf = (t2 >> 5) & 1;
    asm volatile("s_wait_asynccnt 0x0" ::: "memory");  // my DMA chunk landed
    __syncthreads();                                   // everyone's chunks visible; prev iter's reads done
    if (t2 + 32 < T2C) issue_tile(buf ^ 1, t2 + 32);   // overlap next tile's DMA with compute

    // scores: Q(16x64) x K^T(64x32) -> two 16x16 f32 tiles (B-frags from LDS)
    const v16bf bk00 = load_bT_bf(kbuf[buf], 64, 0,  0,  lane);
    const v16bf bk01 = load_bT_bf(kbuf[buf], 64, 0,  32, lane);
    const v16bf bk10 = load_bT_bf(kbuf[buf], 64, 16, 0,  lane);
    const v16bf bk11 = load_bT_bf(kbuf[buf], 64, 16, 32, lane);
    v8f s0 = {}; s0 = wmma_bf16(aq0, bk00, s0); s0 = wmma_bf16(aq1, bk01, s0);
    v8f s1 = {}; s1 = wmma_bf16(aq0, bk10, s1); s1 = wmma_bf16(aq1, bk11, s1);

    const bool ok0 = ma[t2 + m16] != 0;
    const bool ok1 = ma[t2 + 16 + m16] != 0;

    float corr[8];
#pragma unroll
    for (int r = 0; r < 8; ++r) {
      float a0 = ok0 ? s0[r] : -__builtin_inff();
      float a1 = ok1 ? s1[r] : -__builtin_inff();
      float mx = fmaxf(a0, a1);
#pragma unroll
      for (int off = 1; off < 16; off <<= 1)
        mx = fmaxf(mx, __shfl_xor(mx, off, 16));
      const float mnew = fmaxf(rmax[r], mx);     // finite from t2=0 (mask[...,0]==1)
      const float c = __expf(rmax[r] - mnew);
      rmax[r] = mnew; corr[r] = c;
      const float p0 = __expf(a0 - mnew);
      const float p1 = __expf(a1 - mnew);
      float ps = p0 + p1;
#pragma unroll
      for (int off = 1; off < 16; off <<= 1)
        ps += __shfl_xor(ps, off, 16);
      rsum[r] = rsum[r] * c + ps;
      const int m = r + 8 * half;
      plds[wave][m * 32 + m16]      = (__bf16)p0;
      plds[wave][m * 32 + 16 + m16] = (__bf16)p1;
    }
#pragma unroll
    for (int j = 0; j < 4; ++j)
#pragma unroll
      for (int r = 0; r < 8; ++r)
        acc[j][r] *= corr[r];

    // per-wave LDS bounce to re-layout P(16x32) as an A fragment
    asm volatile("s_wait_dscnt 0x0" ::: "memory");
    v16bf pA;
    {
      const __bf16* p = &plds[wave][m16 * 32 + half * 8];
      ((uint4*)&pA)[0] = *(const uint4*)p;
      ((uint4*)&pA)[1] = *(const uint4*)(p + 16);
    }
    // phone += P(16x32) x V(32x64)  (V B-frags from LDS)
#pragma unroll
    for (int j = 0; j < 4; ++j) {
      const v16bf bv = load_bT_bf(vbuf[buf], 32, j * 16, 0, lane);
      acc[j] = wmma_bf16(pA, bv, acc[j]);
    }
  }

  float* ph = phone + (size_t)bnh * T1C * DKC;
#pragma unroll
  for (int r = 0; r < 8; ++r) {
    const int m = r + 8 * half;
    const float inv = 1.f / rsum[r];
#pragma unroll
    for (int j = 0; j < 4; ++j)
      ph[(size_t)(t1_0 + m) * DKC + j * 16 + m16] = acc[j][r] * inv;
  }
}

// ---------------- Stage 2: inter-candidate softmax over N=4 ----------------
__global__ __launch_bounds__(256)
void attn_stage2(const float* __restrict__ qf,     // (B*T1, F), unscaled
                 const float* __restrict__ phone,  // (B*N*H, T1, DK)
                 const int* __restrict__ mask_spk, // (B, N)
                 __bf16* __restrict__ xb)          // (B*T1, F) bf16 for final GEMM
{
  const int idx = blockIdx.x * blockDim.x + threadIdx.x;  // (b, t, h)
  if (idx >= BB * T1C * HH) return;
  const int h = idx & (HH - 1);
  const int t = (idx >> 3) & (T1C - 1);
  const int b = idx >> 11;

  const float* q = qf + (size_t)(b * T1C + t) * FF + h * 64;
  float ss[NN];
#pragma unroll
  for (int n = 0; n < NN; ++n) {
    if (mask_spk[b * NN + n]) {
      const float* ph = phone + (((size_t)(b * NN + n) * HH + h) * T1C + t) * DKC;
      float d = 0.f;
      for (int i = 0; i < DKC; i += 4) {
        const float4 qv = *(const float4*)(q + i);
        const float4 pv = *(const float4*)(ph + i);
        d += qv.x * pv.x + qv.y * pv.y + qv.z * pv.z + qv.w * pv.w;
      }
      ss[n] = d * SCALE_QK;
    } else {
      ss[n] = -__builtin_inff();
    }
  }
  const float mx = fmaxf(fmaxf(ss[0], ss[1]), fmaxf(ss[2], ss[3]));
  float w[NN], sum = 0.f;
#pragma unroll
  for (int n = 0; n < NN; ++n) { w[n] = __expf(ss[n] - mx); sum += w[n]; }
  const float inv = 1.f / sum;

  __bf16* xo = xb + (size_t)(b * T1C + t) * FF + h * 64;
  for (int i = 0; i < DKC; i += 4) {
    float4 o = {0.f, 0.f, 0.f, 0.f};
#pragma unroll
    for (int n = 0; n < NN; ++n) {
      const float4 pv = *(const float4*)(phone +
          (((size_t)(b * NN + n) * HH + h) * T1C + t) * DKC + i);
      const float wn = w[n] * inv;
      o.x += wn * pv.x; o.y += wn * pv.y; o.z += wn * pv.z; o.w += wn * pv.w;
    }
    v4bf ov;
    ov[0] = (__bf16)o.x; ov[1] = (__bf16)o.y; ov[2] = (__bf16)o.z; ov[3] = (__bf16)o.w;
    *(v4bf*)(xo + i) = ov;
  }
}

// ---------------- host launcher ----------------
extern "C" void kernel_launch(void* const* d_in, const int* in_sizes, int n_in,
                              void* d_out, int out_size, void* d_ws, size_t ws_size,
                              hipStream_t stream) {
  const float* query      = (const float*)d_in[0];
  const float* key        = (const float*)d_in[1];
  const float* value      = (const float*)d_in[2];
  const int*   mask_audio = (const int*)d_in[3];
  const int*   mask_spk   = (const int*)d_in[4];
  const float* Wq = (const float*)d_in[5];
  const float* bq = (const float*)d_in[6];
  const float* Wk = (const float*)d_in[7];
  const float* bk = (const float*)d_in[8];
  const float* Wv = (const float*)d_in[9];
  const float* bv = (const float*)d_in[10];
  const float* Wo = (const float*)d_in[11];
  const float* bo = (const float*)d_in[12];

  const int Mq  = BB * T1C;        // 4096
  const int Mkv = BB * NN * T2C;   // 65536
  const int WSZ = FF * FF;         // 262144

  char* ws = (char*)d_ws;
  float*  qf    = (float*)ws;  ws += (size_t)Mq  * FF * 4;  // projected q f32 (stage 2)
  __bf16* qb    = (__bf16*)ws; ws += (size_t)Mq  * FF * 2;  // projected q bf16, pre-scaled
  __bf16* kb    = (__bf16*)ws; ws += (size_t)Mkv * FF * 2;  // projected k bf16
  __bf16* vtb   = (__bf16*)ws; ws += (size_t)Mkv * FF * 2;  // projected v bf16, (bn,h,dk,T2)
  float*  phone = (float*)ws;  ws += (size_t)BB * NN * HH * T1C * DKC * 4;
  __bf16* xb    = (__bf16*)ws; ws += (size_t)Mq  * FF * 2;  // stage-2 output bf16
  __bf16* qin   = (__bf16*)ws; ws += (size_t)Mq  * FF * 2;  // packed inputs
  __bf16* kin   = (__bf16*)ws; ws += (size_t)Mkv * FF * 2;
  __bf16* vin   = (__bf16*)ws; ws += (size_t)Mkv * FF * 2;
  __bf16* wqb   = (__bf16*)ws; ws += (size_t)WSZ * 2;       // packed weights
  __bf16* wkb   = (__bf16*)ws; ws += (size_t)WSZ * 2;
  __bf16* wvb   = (__bf16*)ws; ws += (size_t)WSZ * 2;
  __bf16* wob   = (__bf16*)ws;

  // one-time fp32 -> bf16 packing (removes all cvt work from GEMM inner loops)
  const int nq4 = Mq * FF / 4, nkv4 = Mkv * FF / 4, nw4 = WSZ / 4;
  pack_bf16<<<(nq4  + 255) / 256, 256, 0, stream>>>(query, qin, nq4);
  pack_bf16<<<(nkv4 + 255) / 256, 256, 0, stream>>>(key,   kin, nkv4);
  pack_bf16<<<(nkv4 + 255) / 256, 256, 0, stream>>>(value, vin, nkv4);
  pack_bf16<<<(nw4  + 255) / 256, 256, 0, stream>>>(Wq, wqb, nw4);
  pack_bf16<<<(nw4  + 255) / 256, 256, 0, stream>>>(Wk, wkb, nw4);
  pack_bf16<<<(nw4  + 255) / 256, 256, 0, stream>>>(Wv, wvb, nw4);
  pack_bf16<<<(nw4  + 255) / 256, 256, 0, stream>>>(Wo, wob, nw4);

  // projections (bf16 WMMA; K/V dominate at ~69 GFLOP)
  gemm_bf16_wmma<<<Mq  / 32, 256, 0, stream>>>(qin, wqb, bq, qf,      qb,  SCALE_QK, Mq,  FF, FF, 0);
  gemm_bf16_wmma<<<Mkv / 32, 256, 0, stream>>>(kin, wkb, bk, nullptr, kb,  1.0f,     Mkv, FF, FF, 0);
  gemm_bf16_wmma<<<Mkv / 32, 256, 0, stream>>>(vin, wvb, bv, nullptr, vtb, 1.0f,     Mkv, FF, FF, T2C);

  attn_stage1<<<BB * NN * HH, 512, 0, stream>>>(qb, kb, vtb, mask_audio, phone);

  attn_stage2<<<(BB * T1C * HH + 255) / 256, 256, 0, stream>>>(qf, phone, mask_spk, xb);

  gemm_bf16_wmma<<<Mq / 32, 256, 0, stream>>>(xb, wob, bo, (float*)d_out, nullptr, 1.0f, Mq, FF, FF, 0);
}
